// GCNN_30631706755179
// MI455X (gfx1250) — matrix operand order repaired and checked by reading, not compiled, not run on hardware
//
#include <hip/hip_runtime.h>

// ---------------------------------------------------------------------------
// MI455X / gfx1250 (CDNA5, wave32) implementation.
//  - All three GEMM stages use v_wmma_f32_16x16x32_f16 (f16 A/B, f32 acc).
//  - Activations stored f16 with +1 front / +16 back zero-pad rows so all
//    WMMA A-loads are branch-free (no EXEC churn in the inner loop).
//  - gemm_xlxr: 2 column tiles per wave -> A fragment reuse across 2 WMMAs.
//  - Edge phase: one wave32 per edge, float4 per lane, shfl_xor head reduce.
//  - xl||xr (51MB) is L2-resident (192MB L2) -> gathers are L2-bound.
// ---------------------------------------------------------------------------

typedef __attribute__((ext_vector_type(16))) _Float16     v16h;
typedef __attribute__((ext_vector_type(8)))  float        v8f;
typedef __attribute__((ext_vector_type(4)))  unsigned int u32x4;

#define HD    256   // hidden dim (K for GEMM 1, also output row width)
#define HOUT  128   // H*OUT = HD/2
#define NHEAD 4
#define PADF  1     // zero rows before row 0 in padded f16 activations
#define PADB  16    // zero rows after row N-1

// ---- ordered-uint encoding so we can do float max via atomicMax(u32) ------
__device__ __forceinline__ unsigned ordEnc(float f) {
  unsigned u = __float_as_uint(f);
  return (u & 0x80000000u) ? ~u : (u | 0x80000000u);
}
__device__ __forceinline__ float ordDec(unsigned u) {
  return __uint_as_float((u & 0x80000000u) ? (u & 0x7fffffffu) : ~u);
}
#define ORD_NEG_INF 0x007FFFFFu   // ordEnc(-inf)

// ---- WMMA fragment loaders (layouts per CDNA5 ISA 7.12.2) -----------------
// A 16x32 f16, row-major source [rows][ldk] (caller guarantees rows valid):
//   lanes 0-15 : M=lane,    elems 0..7 = K 0..7,  elems 8..15 = K 16..23
//   lanes 16-31: M=lane-16, elems 0..7 = K 8..15, elems 8..15 = K 24..31
__device__ __forceinline__ v16h load_a16x32(const _Float16* base, int rowTile,
                                            int k0, int lane, int ldk) {
  int r  = rowTile + (lane & 15);
  int lo = (lane >> 4) << 3;                            // 0 or 8
  const _Float16* p = base + (size_t)r * ldk + k0 + lo;
  union { u32x4 q[2]; v16h v; } f;
  f.q[0] = *(const u32x4*)(p);        // K = k0+lo    .. +7   (16B load)
  f.q[1] = *(const u32x4*)(p + 16);   // K = k0+lo+16 .. +7   (16B load)
  return f.v;
}
// B 32x16 f16 from PRE-TRANSPOSED weights Wt[Ncol][ldk]:
//   lane 0-15: K = k0+0..7, k0+16..23; lane 16-31: K = k0+8..15, k0+24..31;  N = lane&15
__device__ __forceinline__ v16h load_b32x16(const _Float16* baseT, int colTile,
                                            int k0, int lane, int ldk) {
  int n  = colTile + (lane & 15);
  int kk = k0 + ((lane >> 4) << 3);                     // +0 or +8
  const _Float16* p = baseT + (size_t)n * ldk + kk;
  union { u32x4 q[2]; v16h v; } f;
  f.q[0] = *(const u32x4*)(p);
  f.q[1] = *(const u32x4*)(p + 16);                     // K = kk+16 .. +7
  return f.v;
}

// ---------------------------------------------------------------------------
// Casting / layout prep
// ---------------------------------------------------------------------------
__global__ void cast_f16(const float* __restrict__ src, _Float16* __restrict__ dst, long n) {
  long i = (long)blockIdx.x * blockDim.x + threadIdx.x;
  if (i < n) dst[i] = (_Float16)src[i];
}

// WlrT[c][k] = (c<128 ? W_l[k][c] : W_r[k][c-128]),  shape [256][256]
__global__ void cast_wlr(const float* __restrict__ Wl, const float* __restrict__ Wr,
                         _Float16* __restrict__ wlrT) {
  int i = blockIdx.x * blockDim.x + threadIdx.x;
  if (i >= HD * HD) return;
  int c = i / HD, k = i % HD;
  float v = (c < HOUT) ? Wl[k * HOUT + c] : Wr[k * HOUT + (c - HOUT)];
  wlrT[i] = (_Float16)v;
}

// WkT[kk][o][i] = conv_w[o][i][kk]   (conv_w is [128][256][3] row-major)
__global__ void cast_wk(const float* __restrict__ conv_w, _Float16* __restrict__ wkT) {
  int i = blockIdx.x * blockDim.x + threadIdx.x;
  if (i >= 3 * HOUT * HD) return;
  int kk  = i / (HOUT * HD);
  int rem = i % (HOUT * HD);
  int o = rem / HD, ci = rem % HD;
  wkT[i] = (_Float16)conv_w[(size_t)o * (HD * 3) + ci * 3 + kk];
}

// zero accumulators + softmax state + pad rows of the f16 activation buffer
__global__ void init_buf(float* x1, float* denom, unsigned* mOrd,
                         float* bnsum, float* bnsq, _Float16* xbPad, int N) {
  size_t i = (size_t)blockIdx.x * blockDim.x + threadIdx.x;
  if (i < (size_t)N * HOUT)  x1[i] = 0.f;
  if (i < (size_t)N * NHEAD) { denom[i] = 0.f; mOrd[i] = ORD_NEG_INF; }
  if (i < HOUT) { bnsum[i] = 0.f; bnsq[i] = 0.f; }
  if (i < (size_t)(PADF + PADB) * HD) {
    size_t idx = (i < PADF * HD) ? i : ((size_t)(PADF + N) * HD + (i - PADF * HD));
    xbPad[idx] = (_Float16)0.f;
  }
}

// ---------------------------------------------------------------------------
// GEMM 1: xl||xr = x @ [W_l | W_r]   ->  xlxr [N][256] f32
// block = 8 waves; each wave: 16 rows x 32 cols (2 tiles, shared A fragment)
// ---------------------------------------------------------------------------
__global__ void gemm_xlxr(const _Float16* __restrict__ xbPad, const _Float16* __restrict__ wlrT,
                          float* __restrict__ y, int N) {
  const _Float16* xb = xbPad + (size_t)PADF * HD;   // logical row 0
  int wave = threadIdx.x >> 5, lane = threadIdx.x & 31;
  int rowTile  = blockIdx.x * 16;
  int colTile0 = wave * 32;                         // 8 waves * 32 = 256 cols
  v8f acc0 = {}, acc1 = {};
#pragma unroll
  for (int k0 = 0; k0 < HD; k0 += 32) {
    v16h a  = load_a16x32(xb, rowTile, k0, lane, HD);
    v16h b0 = load_b32x16(wlrT, colTile0,      k0, lane, HD);
    v16h b1 = load_b32x16(wlrT, colTile0 + 16, k0, lane, HD);
    acc0 = __builtin_amdgcn_wmma_f32_16x16x32_f16(false, a, false, b0, (short)0, acc0, false, false);
    acc1 = __builtin_amdgcn_wmma_f32_16x16x32_f16(false, a, false, b1, (short)0, acc1, false, false);
  }
  int col   = colTile0 + (lane & 15);
  int rbase = rowTile + ((lane >> 4) << 3);         // D layout: M = r + 8*(lane>=16)
#pragma unroll
  for (int r = 0; r < 8; ++r) {
    int row = rbase + r;
    if (row < N) {
      y[(size_t)row * HD + col]      = acc0[r];
      y[(size_t)row * HD + col + 16] = acc1[r];
    }
  }
}

// ---------------------------------------------------------------------------
// Edge pass A: scores[e][h] = <leaky0.2(xl[src]+xr[dst]), att[h]> ; seg-max
// ---------------------------------------------------------------------------
__global__ void edge_scores(const float* __restrict__ xlxr, const long long* __restrict__ ed,
                            const float* __restrict__ att, float* __restrict__ scores,
                            unsigned* __restrict__ mOrd, int E) {
  int wave = threadIdx.x >> 5, lane = threadIdx.x & 31;
  long e = (long)blockIdx.x * 8 + wave;
  if (e >= E) return;
  int src = (int)ed[e];
  int dst = (int)ed[(size_t)E + e];
  float4 a  = *(const float4*)(xlxr + (size_t)src * HD + lane * 4);
  float4 b  = *(const float4*)(xlxr + (size_t)dst * HD + HOUT + lane * 4);
  float4 ww = *(const float4*)(att + lane * 4);     // att flat: j = h*32+o
  float g0 = a.x + b.x; g0 = g0 > 0.f ? g0 : 0.2f * g0;
  float g1 = a.y + b.y; g1 = g1 > 0.f ? g1 : 0.2f * g1;
  float g2 = a.z + b.z; g2 = g2 > 0.f ? g2 : 0.2f * g2;
  float g3 = a.w + b.w; g3 = g3 > 0.f ? g3 : 0.2f * g3;
  float p = g0 * ww.x + g1 * ww.y + g2 * ww.z + g3 * ww.w;
  p += __shfl_xor(p, 1, 32);    // reduce within 8-lane head group
  p += __shfl_xor(p, 2, 32);
  p += __shfl_xor(p, 4, 32);
  if ((lane & 7) == 0) {
    int h = lane >> 3;
    scores[e * 4 + h] = p;
    atomicMax(mOrd + (size_t)dst * 4 + h, ordEnc(p));
  }
}

// ---------------------------------------------------------------------------
// Edge pass B: ex = exp(score - m[dst]); denom[dst] += ex
// ---------------------------------------------------------------------------
__global__ void edge_exp(const long long* __restrict__ ed, const unsigned* __restrict__ mOrd,
                         float* __restrict__ scores, float* __restrict__ denom, int E) {
  long i = (long)blockIdx.x * blockDim.x + threadIdx.x;
  if (i >= (long)E * 4) return;
  long e = i >> 2; int h = (int)(i & 3);
  int dst = (int)ed[(size_t)E + e];
  float m  = ordDec(mOrd[(size_t)dst * 4 + h]);
  float ex = __expf(scores[i] - m);
  scores[i] = ex;
  atomicAdd(denom + (size_t)dst * 4 + h, ex);
}

// ---------------------------------------------------------------------------
// Edge pass C: alpha = ex/(denom+1e-16); x1[dst] += xl[src]*alpha; emit alpha
// ---------------------------------------------------------------------------
__global__ void edge_scatter(const float* __restrict__ xlxr, const long long* __restrict__ ed,
                             const float* __restrict__ ex, const float* __restrict__ denom,
                             float* __restrict__ x1, float* __restrict__ alphaOut, int E) {
  int wave = threadIdx.x >> 5, lane = threadIdx.x & 31;
  long e = (long)blockIdx.x * 8 + wave;
  if (e >= E) return;
  int src = (int)ed[e];
  int dst = (int)ed[(size_t)E + e];
  int h = lane >> 3;                                   // 4 elems/lane -> head = lane/8
  float al = ex[e * 4 + h] / (denom[(size_t)dst * 4 + h] + 1e-16f);
  if ((lane & 7) == 0) alphaOut[e * 4 + h] = al;
  float4 xs = *(const float4*)(xlxr + (size_t)src * HD + lane * 4);
  float* p = x1 + (size_t)dst * HOUT + lane * 4;
  atomicAdd(p + 0, xs.x * al);
  atomicAdd(p + 1, xs.y * al);
  atomicAdd(p + 2, xs.z * al);
  atomicAdd(p + 3, xs.w * al);
}

// ---------------------------------------------------------------------------
// BatchNorm stats (column sums over N) + apply (+ leaky 0.01) -> out[:, :128]
// ---------------------------------------------------------------------------
__global__ void bn_stats(const float* __restrict__ x1, const float* __restrict__ gbias,
                         float* __restrict__ bnsum, float* __restrict__ bnsq, int N) {
  int c = threadIdx.x;                 // 128 threads = 128 channels
  int n0 = blockIdx.x * 128;
  int nEnd = min(n0 + 128, N);
  float gb = gbias[c];
  float s = 0.f, q = 0.f;
  for (int n = n0; n < nEnd; ++n) {
    float v = x1[(size_t)n * HOUT + c] + gb;
    s += v; q += v * v;
  }
  atomicAdd(&bnsum[c], s);
  atomicAdd(&bnsq[c], q);
}

__global__ void bn_apply(const float* __restrict__ x1, const float* __restrict__ gbias,
                         const float* __restrict__ bnsum, const float* __restrict__ bnsq,
                         const float* __restrict__ gamma, const float* __restrict__ beta,
                         float* __restrict__ out, int N) {
  size_t i = (size_t)blockIdx.x * blockDim.x + threadIdx.x;
  if (i >= (size_t)N * HOUT) return;
  int n = (int)(i / HOUT), c = (int)(i % HOUT);
  float invN = 1.0f / (float)N;
  float mu  = bnsum[c] * invN;
  float var = bnsq[c] * invN - mu * mu;   // biased variance (ddof=0)
  float v = x1[i] + gbias[c];
  float y = (v - mu) * rsqrtf(var + 1e-5f) * gamma[c] + beta[c];
  y = y > 0.f ? y : 0.01f * y;
  out[(size_t)n * HD + c] = y;
}

// ---------------------------------------------------------------------------
// Conv1d over node axis = 3 shifted WMMA GEMM accumulations -> out[:, 128:]
// Padded activations make the kk-1 row shift a plain (branch-free) offset.
// ---------------------------------------------------------------------------
__global__ void conv_gemm(const _Float16* __restrict__ xbPad, const _Float16* __restrict__ wkT,
                          const float* __restrict__ cb, float* __restrict__ out, int N) {
  int wave = threadIdx.x >> 5, lane = threadIdx.x & 31;
  int rowTile = blockIdx.x * 16;
  int colTile = wave * 16;                     // 8 waves cover 128 output cols
  v8f acc = {};
  for (int kk = 0; kk < 3; ++kk) {
    // logical A row = rowTile + (kk-1); +PADF pad shift => rowTile + kk
    const _Float16* abase = xbPad + (size_t)(0) * HD;
    const _Float16* wt = wkT + (size_t)kk * HOUT * HD;
#pragma unroll
    for (int k0 = 0; k0 < HD; k0 += 32) {
      v16h a = load_a16x32(abase, rowTile + kk, k0, lane, HD);
      v16h b = load_b32x16(wt, colTile, k0, lane, HD);
      acc = __builtin_amdgcn_wmma_f32_16x16x32_f16(false, a, false, b, (short)0, acc, false, false);
    }
  }
  int col   = colTile + (lane & 15);
  int rbase = rowTile + ((lane >> 4) << 3);
  float bias = cb[col];
#pragma unroll
  for (int r = 0; r < 8; ++r) {
    int row = rbase + r;
    if (row < N) {
      float v = acc[r] + bias;
      out[(size_t)row * HD + HOUT + col] = v > 0.f ? v : 0.f;   // ReLU
    }
  }
}

// edge_data passthrough into output tail (values < 50000: exact in f32)
__global__ void copy_edges(const long long* __restrict__ ed, float* __restrict__ dst, long n) {
  long i = (long)blockIdx.x * blockDim.x + threadIdx.x;
  if (i < n) dst[i] = (float)ed[i];
}

// ---------------------------------------------------------------------------
extern "C" void kernel_launch(void* const* d_in, const int* in_sizes, int n_in,
                              void* d_out, int out_size, void* d_ws, size_t ws_size,
                              hipStream_t stream) {
  const float*     x      = (const float*)d_in[0];
  const long long* ed     = (const long long*)d_in[1];
  const float*     W_l    = (const float*)d_in[2];
  const float*     W_r    = (const float*)d_in[3];
  const float*     att    = (const float*)d_in[4];
  const float*     gbias  = (const float*)d_in[5];
  const float*     conv_w = (const float*)d_in[6];
  const float*     conv_b = (const float*)d_in[7];
  const float*     gamma  = (const float*)d_in[8];
  const float*     beta   = (const float*)d_in[9];
  const int N = in_sizes[0] / HD;    // 50000
  const int E = in_sizes[1] / 2;     // 800000
  float* out = (float*)d_out;

  // ---- workspace carve (aligned 256B); total ~117 MB ----
  char* w = (char*)d_ws;
  auto carve = [&](size_t bytes) -> void* {
    void* p = (void*)w; w += (bytes + 255) & ~(size_t)255; return p;
  };
  _Float16* xbPad = (_Float16*)carve((size_t)(N + PADF + PADB) * HD * sizeof(_Float16));
  _Float16* wlrT  = (_Float16*)carve((size_t)HD * HD * sizeof(_Float16));
  _Float16* wkT   = (_Float16*)carve((size_t)3 * HOUT * HD * sizeof(_Float16));
  float*    xlxr  = (float*)carve((size_t)N * HD * sizeof(float));
  float*    scores= (float*)carve((size_t)E * NHEAD * sizeof(float));
  unsigned* mOrd  = (unsigned*)carve((size_t)N * NHEAD * sizeof(unsigned));
  float*    denom = (float*)carve((size_t)N * NHEAD * sizeof(float));
  float*    x1    = (float*)carve((size_t)N * HOUT * sizeof(float));
  float*    bnsum = (float*)carve(HOUT * sizeof(float));
  float*    bnsq  = (float*)carve(HOUT * sizeof(float));

  float* alphaOut = out + (size_t)N * HD;
  float* edgeOut  = alphaOut + (size_t)E * NHEAD;

  // ---- prep ----
  { long n = (long)N * HD;   // fill logical rows [0,N) of the padded buffer
    cast_f16<<<(unsigned)((n + 255) / 256), 256, 0, stream>>>(x, xbPad + (size_t)PADF * HD, n); }
  cast_wlr<<<(HD * HD + 255) / 256, 256, 0, stream>>>(W_l, W_r, wlrT);
  cast_wk <<<(3 * HOUT * HD + 255) / 256, 256, 0, stream>>>(conv_w, wkT);
  { size_t n = (size_t)N * HOUT;
    init_buf<<<(unsigned)((n + 255) / 256), 256, 0, stream>>>(x1, denom, mOrd, bnsum, bnsq,
                                                              xbPad, N); }

  // ---- GATv2 ----
  gemm_xlxr<<<(N + 15) / 16, 256, 0, stream>>>(xbPad, wlrT, xlxr, N);
  edge_scores<<<(E + 7) / 8, 256, 0, stream>>>(xlxr, ed, att, scores, mOrd, E);
  { long n = (long)E * NHEAD;
    edge_exp<<<(unsigned)((n + 255) / 256), 256, 0, stream>>>(ed, mOrd, scores, denom, E); }
  edge_scatter<<<(E + 7) / 8, 256, 0, stream>>>(xlxr, ed, scores, denom, x1, alphaOut, E);

  // ---- BatchNorm + LeakyReLU -> out[:, :128] ----
  bn_stats<<<(N + 127) / 128, 128, 0, stream>>>(x1, gbias, bnsum, bnsq, N);
  { size_t n = (size_t)N * HOUT;
    bn_apply<<<(unsigned)((n + 255) / 256), 256, 0, stream>>>(x1, gbias, bnsum, bnsq,
                                                              gamma, beta, out, N); }

  // ---- Conv1d + ReLU -> out[:, 128:] ----
  conv_gemm<<<(N + 15) / 16, 256, 0, stream>>>(xbPad, wkT, conv_b, out, N);

  // ---- edge_data passthrough ----
  { long n = 2L * E;
    copy_edges<<<(unsigned)((n + 255) / 256), 256, 0, stream>>>(ed, edgeOut, n); }
}
